// EGNNLayer_78391743086889
// MI455X (gfx1250) — matrix operand order repaired and checked by reading, not compiled, not run on hardware
//
#include <hip/hip_runtime.h>

typedef __attribute__((ext_vector_type(16))) _Float16 v16h;
typedef __attribute__((ext_vector_type(8)))  _Float16 v8h;
typedef __attribute__((ext_vector_type(8)))  float    v8f;

__device__ __forceinline__ float siluf(float v) {
    return v / (1.0f + __expf(-v));
}

__device__ __forceinline__ v8f wmma_f16(v16h a, v16h b, v8f c) {
    return __builtin_amdgcn_wmma_f32_16x16x32_f16(false, a, false, b, (short)0, c, false, false);
}

// A-tile (16x32 f16) load: row = lane&15, kb = (lane&16)?8:0.
// Elements i=0..7 -> K=kb+i ; i=8..15 -> K=kb+16+(i-8). Two contiguous 16B chunks.
__device__ __forceinline__ v16h load_tileA(const _Float16* rowptr, int kb) {
    v8h lo = *(const v8h*)(rowptr + kb);
    v8h hi = *(const v8h*)(rowptr + kb + 16);
    return __builtin_shufflevector(lo, hi, 0,1,2,3,4,5,6,7,8,9,10,11,12,13,14,15);
}

// B-tile (32x16 f16) from pre-swizzled storage: 512 f16 per tile, lane-major 16 each.
__device__ __forceinline__ v16h load_tileB(const _Float16* sw, int tile, int lane) {
    const _Float16* p = sw + tile * 512 + lane * 16;
    v8h lo = *(const v8h*)(p);
    v8h hi = *(const v8h*)(p + 8);
    return __builtin_shufflevector(lo, hi, 0,1,2,3,4,5,6,7,8,9,10,11,12,13,14,15);
}

// ---------------- prep kernels ----------------

__global__ void zero_f32_kernel(float* p, int n) {
    int i = blockIdx.x * blockDim.x + threadIdx.x;
    if (i < n) p[i] = 0.0f;
}

__global__ void cvt_f32_to_f16_kernel(const float* __restrict__ s, _Float16* __restrict__ d, int n) {
    int i = blockIdx.x * blockDim.x + threadIdx.x;
    if (i < n) d[i] = (_Float16)s[i];
}

// Swizzle row-major f32 W[K][Ncols] into WMMA B-operand tile layout (f16).
// tile = kt*(Ncols/16)+nt ; lane n=lane&15, kbase=(lane&16)?8:0, element i<8 -> K=kt*32+kbase+i ; i>=8 -> K=kt*32+kbase+16+(i-8).
__global__ void swizzleB_kernel(const float* __restrict__ src, _Float16* __restrict__ dst,
                                int K, int Ncols) {
    int tile = blockIdx.x;
    int lane = threadIdx.x;
    int ntn  = Ncols >> 4;
    int kt = tile / ntn, nt = tile % ntn;
    int n  = nt * 16 + (lane & 15);
    int k0 = kt * 32 + ((lane & 16) ? 8 : 0);
    _Float16* dp = dst + tile * 512 + lane * 16;
#pragma unroll
    for (int i = 0; i < 16; ++i)
        dp[i] = (_Float16)src[(size_t)(k0 + i + (i >= 8 ? 8 : 0)) * Ncols + n];
}

// ---------------- edge kernel ----------------
// blocks of 64 threads (2 waves); each wave does 8 tiles of 16 edges (256 edges/block).

__global__ __launch_bounds__(64) void egnn_edge_kernel(
    const _Float16* __restrict__ h16, const float* __restrict__ x,
    const int* __restrict__ src, const int* __restrict__ dst,
    const _Float16* __restrict__ w1sw, const _Float16* __restrict__ w2sw,
    const float* __restrict__ ew1,  // for last row (x_diff_sq weight)
    const float* __restrict__ eb1, const float* __restrict__ eb2,
    const float* __restrict__ cw1, const float* __restrict__ cb1,
    const float* __restrict__ cw2, const float* __restrict__ cb2,
    const float* __restrict__ cscal,
    float* __restrict__ m_i, float* __restrict__ v_upd,
    int E, int N)
{
    __shared__ __align__(16) _Float16 sW1[128 * 128];   // 32 KB (ew1 rows 0..127, swizzled)
    __shared__ __align__(16) _Float16 sW2[128 * 64];    // 16 KB
    __shared__ float sW1last[128];
    __shared__ float sEB1[128];
    __shared__ float sEB2[64];
    __shared__ float sCW1[64];
    __shared__ __align__(16) _Float16 sHid[2][16 * 128]; // 8 KB bounce buffers
    __shared__ float sXd[2][16][3];
    __shared__ float sXds[2][16];
    __shared__ int   sSrc[2][16];
    __shared__ int   sDst[2][16];

    const int tid = threadIdx.x;
    // cooperative weight load (16B chunks)
    for (int i = tid; i < (128 * 128 * 2) / 16; i += 64)
        ((uint4*)sW1)[i] = ((const uint4*)w1sw)[i];
    for (int i = tid; i < (128 * 64 * 2) / 16; i += 64)
        ((uint4*)sW2)[i] = ((const uint4*)w2sw)[i];
    for (int i = tid; i < 128; i += 64) {
        sW1last[i] = ew1[128 * 128 + i];
        sEB1[i]    = eb1[i];
    }
    if (tid < 64) { sEB2[tid] = eb2[tid]; sCW1[tid] = cw1[tid]; }
    __syncthreads();

    const int wvi  = tid >> 5;
    const int lane = tid & 31;
    const int rA   = lane & 15;
    const int kbA  = (lane & 16) ? 8 : 0;
    const int cB   = lane & 15;
    const int rOff = (lane & 16) ? 8 : 0;

    for (int t = 0; t < 8; ++t) {
        const int e0 = blockIdx.x * 256 + wvi * 128 + t * 16;
        if (lane < 16) {
            int e = e0 + lane; if (e >= E) e = E - 1;
            int s = src[e], d = dst[e];
            sSrc[wvi][lane] = s;
            sDst[wvi][lane] = d;
            float d0 = x[(size_t)d * 3 + 0] - x[(size_t)s * 3 + 0];
            float d1 = x[(size_t)d * 3 + 1] - x[(size_t)s * 3 + 1];
            float d2 = x[(size_t)d * 3 + 2] - x[(size_t)s * 3 + 2];
            sXd[wvi][lane][0] = d0; sXd[wvi][lane][1] = d1; sXd[wvi][lane][2] = d2;
            sXds[wvi][lane] = d0 * d0 + d1 * d1 + d2 * d2;
        }
        __syncthreads();

        const int sI = sSrc[wvi][rA];
        const int dI = sDst[wvi][rA];
        v16h A0 = load_tileA(h16 + (size_t)sI * 64,      kbA);
        v16h A1 = load_tileA(h16 + (size_t)sI * 64 + 32, kbA);
        v16h A2 = load_tileA(h16 + (size_t)dI * 64,      kbA);
        v16h A3 = load_tileA(h16 + (size_t)dI * 64 + 32, kbA);

        float xv[8];
#pragma unroll
        for (int r = 0; r < 8; ++r) xv[r] = sXds[wvi][r + rOff];

        // GEMM1: hidden[16,128] = [h_src|h_dst]@W1 + xds*w1_last + eb1, then SiLU -> f16 LDS
#pragma unroll
        for (int nt = 0; nt < 8; ++nt) {
            v8f acc = {};
            acc = wmma_f16(A0, load_tileB(sW1, 0 * 8 + nt, lane), acc);
            acc = wmma_f16(A1, load_tileB(sW1, 1 * 8 + nt, lane), acc);
            acc = wmma_f16(A2, load_tileB(sW1, 2 * 8 + nt, lane), acc);
            acc = wmma_f16(A3, load_tileB(sW1, 3 * 8 + nt, lane), acc);
            const int col = nt * 16 + cB;
            const float b1 = sEB1[col], wl = sW1last[col];
#pragma unroll
            for (int r = 0; r < 8; ++r) {
                float v = acc[r] + b1 + xv[r] * wl;
                sHid[wvi][(r + rOff) * 128 + col] = (_Float16)siluf(v);
            }
        }
        __syncthreads();

        // GEMM2: m_ij[16,64] = silu(hid@W2 + eb2); scatter + coord dot
        const _Float16* hrow = &sHid[wvi][rA * 128];
        v16h H0 = load_tileA(hrow,      kbA);
        v16h H1 = load_tileA(hrow + 32, kbA);
        v16h H2 = load_tileA(hrow + 64, kbA);
        v16h H3 = load_tileA(hrow + 96, kbA);

        float p[8] = {0, 0, 0, 0, 0, 0, 0, 0};
#pragma unroll
        for (int nt = 0; nt < 4; ++nt) {
            v8f acc = {};
            acc = wmma_f16(H0, load_tileB(sW2, 0 * 4 + nt, lane), acc);
            acc = wmma_f16(H1, load_tileB(sW2, 1 * 4 + nt, lane), acc);
            acc = wmma_f16(H2, load_tileB(sW2, 2 * 4 + nt, lane), acc);
            acc = wmma_f16(H3, load_tileB(sW2, 3 * 4 + nt, lane), acc);
            const int col = nt * 16 + cB;
            const float b2 = sEB2[col], c1 = sCW1[col];
#pragma unroll
            for (int r = 0; r < 8; ++r) {
                float mv = siluf(acc[r] + b2);
                p[r] += mv * c1;
                int e = e0 + r + rOff;
                if (e < E)
                    atomicAdd(&m_i[(size_t)sDst[wvi][r + rOff] * 64 + col], mv);
            }
        }
        // reduce coord dot across the 16 lanes of each half-wave
#pragma unroll
        for (int off = 8; off; off >>= 1)
#pragma unroll
            for (int r = 0; r < 8; ++r)
                p[r] += __shfl_xor(p[r], off, 16);

        if ((lane & 15) < 8) {
            int r = lane & 7;
            int m = r + rOff;
            int e = e0 + m;
            if (e < E) {
                float w  = siluf(p[r] + cb1[0]) * cw2[0] + cb2[0];
                float cm = cscal[0] * w;
                float* vp = v_upd + (size_t)sDst[wvi][m] * 3;
                atomicAdd(vp + 0, cm * sXd[wvi][m][0]);
                atomicAdd(vp + 1, cm * sXd[wvi][m][1]);
                atomicAdd(vp + 2, cm * sXd[wvi][m][2]);
            }
        }
    }
}

// ---------------- node h-path kernel ----------------

__global__ __launch_bounds__(64) void egnn_node_h_kernel(
    const _Float16* __restrict__ h16, const float* __restrict__ m_i,
    const float* __restrict__ h,
    const _Float16* __restrict__ nw1sw, const _Float16* __restrict__ nw2sw,
    const float* __restrict__ nb1, const float* __restrict__ nb2,
    float* __restrict__ out_h, int N)
{
    __shared__ __align__(16) _Float16 sW1[128 * 128];
    __shared__ __align__(16) _Float16 sW2[128 * 64];
    __shared__ float sNB1[128];
    __shared__ float sNB2[64];
    __shared__ __align__(16) _Float16 sHid[2][16 * 128];

    const int tid = threadIdx.x;
    for (int i = tid; i < (128 * 128 * 2) / 16; i += 64)
        ((uint4*)sW1)[i] = ((const uint4*)nw1sw)[i];
    for (int i = tid; i < (128 * 64 * 2) / 16; i += 64)
        ((uint4*)sW2)[i] = ((const uint4*)nw2sw)[i];
    for (int i = tid; i < 128; i += 64) sNB1[i] = nb1[i];
    if (tid < 64) sNB2[tid] = nb2[tid];
    __syncthreads();

    const int wvi  = tid >> 5;
    const int lane = tid & 31;
    const int rA   = lane & 15;
    const int kbA  = (lane & 16) ? 8 : 0;
    const int cB   = lane & 15;
    const int rOff = (lane & 16) ? 8 : 0;

    for (int t = 0; t < 8; ++t) {
        const int n0 = blockIdx.x * 256 + wvi * 128 + t * 16;
        int nd = n0 + rA; if (nd >= N) nd = N - 1;

        // h_input = [h(64) | m_i(64)]
        v16h A0 = load_tileA(h16 + (size_t)nd * 64,      kbA);
        v16h A1 = load_tileA(h16 + (size_t)nd * 64 + 32, kbA);
        v16h A2, A3;
        {
            const float* pm = m_i + (size_t)nd * 64;
#pragma unroll
            for (int i = 0; i < 8; ++i) {
                A2[i]     = (_Float16)pm[kbA + i];
                A2[8 + i] = (_Float16)pm[kbA + 16 + i];
                A3[i]     = (_Float16)pm[32 + kbA + i];
                A3[8 + i] = (_Float16)pm[32 + kbA + 16 + i];
            }
        }

#pragma unroll
        for (int nt = 0; nt < 8; ++nt) {
            v8f acc = {};
            acc = wmma_f16(A0, load_tileB(sW1, 0 * 8 + nt, lane), acc);
            acc = wmma_f16(A1, load_tileB(sW1, 1 * 8 + nt, lane), acc);
            acc = wmma_f16(A2, load_tileB(sW1, 2 * 8 + nt, lane), acc);
            acc = wmma_f16(A3, load_tileB(sW1, 3 * 8 + nt, lane), acc);
            const int col = nt * 16 + cB;
            const float b1 = sNB1[col];
#pragma unroll
            for (int r = 0; r < 8; ++r)
                sHid[wvi][(r + rOff) * 128 + col] = (_Float16)siluf(acc[r] + b1);
        }
        __syncthreads();

        const _Float16* hrow = &sHid[wvi][rA * 128];
        v16h H0 = load_tileA(hrow,      kbA);
        v16h H1 = load_tileA(hrow + 32, kbA);
        v16h H2 = load_tileA(hrow + 64, kbA);
        v16h H3 = load_tileA(hrow + 96, kbA);

#pragma unroll
        for (int nt = 0; nt < 4; ++nt) {
            v8f acc = {};
            acc = wmma_f16(H0, load_tileB(sW2, 0 * 4 + nt, lane), acc);
            acc = wmma_f16(H1, load_tileB(sW2, 1 * 4 + nt, lane), acc);
            acc = wmma_f16(H2, load_tileB(sW2, 2 * 4 + nt, lane), acc);
            acc = wmma_f16(H3, load_tileB(sW2, 3 * 4 + nt, lane), acc);
            const int col = nt * 16 + cB;
            const float b2 = sNB2[col];
#pragma unroll
            for (int r = 0; r < 8; ++r) {
                int node = n0 + r + rOff;
                if (node < N)
                    out_h[(size_t)node * 64 + col] =
                        acc[r] + b2 + h[(size_t)node * 64 + col];
            }
        }
        __syncthreads();
    }
}

// ---------------- node v-path kernel ----------------

__global__ __launch_bounds__(64) void egnn_node_v_kernel(
    const _Float16* __restrict__ h16,
    const float* __restrict__ x, const float* __restrict__ v_init,
    const float* __restrict__ vw2, const float* __restrict__ vb2,
    const _Float16* __restrict__ vw1sw, const float* __restrict__ vb1,
    const float* __restrict__ v_upd,
    float* __restrict__ out, int N)
{
    __shared__ __align__(16) _Float16 sW1[64 * 128]; // 16 KB
    __shared__ float sVB1[128];

    const int tid = threadIdx.x;
    for (int i = tid; i < (64 * 128 * 2) / 16; i += 64)
        ((uint4*)sW1)[i] = ((const uint4*)vw1sw)[i];
    for (int i = tid; i < 128; i += 64) sVB1[i] = vb1[i];
    __syncthreads();

    const int wvi  = tid >> 5;
    const int lane = tid & 31;
    const int rA   = lane & 15;
    const int kbA  = (lane & 16) ? 8 : 0;
    const int cB   = lane & 15;
    const int rOff = (lane & 16) ? 8 : 0;

    const size_t oX = (size_t)N * 64;       // x_new block in d_out
    const size_t oV = (size_t)N * 67;       // v_new block in d_out

    for (int t = 0; t < 8; ++t) {
        const int n0 = blockIdx.x * 256 + wvi * 128 + t * 16;
        int nd = n0 + rA; if (nd >= N) nd = N - 1;

        v16h A0 = load_tileA(h16 + (size_t)nd * 64,      kbA);
        v16h A1 = load_tileA(h16 + (size_t)nd * 64 + 32, kbA);

        float p[8][3];
#pragma unroll
        for (int r = 0; r < 8; ++r) { p[r][0] = 0.f; p[r][1] = 0.f; p[r][2] = 0.f; }

#pragma unroll
        for (int nt = 0; nt < 8; ++nt) {
            v8f acc = {};
            acc = wmma_f16(A0, load_tileB(sW1, 0 * 8 + nt, lane), acc);
            acc = wmma_f16(A1, load_tileB(sW1, 1 * 8 + nt, lane), acc);
            const int col = nt * 16 + cB;
            const float b  = sVB1[col];
            const float w0 = vw2[(size_t)col * 3 + 0];
            const float w1 = vw2[(size_t)col * 3 + 1];
            const float w2 = vw2[(size_t)col * 3 + 2];
#pragma unroll
            for (int r = 0; r < 8; ++r) {
                float hv = siluf(acc[r] + b);
                p[r][0] += hv * w0;
                p[r][1] += hv * w1;
                p[r][2] += hv * w2;
            }
        }
#pragma unroll
        for (int off = 8; off; off >>= 1)
#pragma unroll
            for (int r = 0; r < 8; ++r) {
                p[r][0] += __shfl_xor(p[r][0], off, 16);
                p[r][1] += __shfl_xor(p[r][1], off, 16);
                p[r][2] += __shfl_xor(p[r][2], off, 16);
            }

        if ((lane & 15) < 8) {
            int r = lane & 7;
            int node = n0 + r + rOff;
            if (node < N) {
#pragma unroll
                for (int j = 0; j < 3; ++j) {
                    float vn = (p[r][j] + vb2[j]) * v_init[(size_t)node * 3 + j]
                             + v_upd[(size_t)node * 3 + j];
                    out[oX + (size_t)node * 3 + j] = x[(size_t)node * 3 + j] + vn;
                    out[oV + (size_t)node * 3 + j] = vn;
                }
            }
        }
    }
}

// ---------------- launch ----------------

static inline size_t align256(size_t v) { return (v + 255) & ~(size_t)255; }

extern "C" void kernel_launch(void* const* d_in, const int* in_sizes, int n_in,
                              void* d_out, int out_size, void* d_ws, size_t ws_size,
                              hipStream_t stream) {
    const float* h      = (const float*)d_in[0];
    const float* x      = (const float*)d_in[1];
    const float* v_init = (const float*)d_in[2];
    const int*   eidx   = (const int*)  d_in[3];
    const float* ew1    = (const float*)d_in[4];
    const float* eb1    = (const float*)d_in[5];
    const float* ew2    = (const float*)d_in[6];
    const float* eb2    = (const float*)d_in[7];
    const float* cw1    = (const float*)d_in[8];
    const float* cb1    = (const float*)d_in[9];
    const float* cw2    = (const float*)d_in[10];
    const float* cb2    = (const float*)d_in[11];
    const float* vw1    = (const float*)d_in[12];
    const float* vb1    = (const float*)d_in[13];
    const float* vw2    = (const float*)d_in[14];
    const float* vb2    = (const float*)d_in[15];
    const float* nw1    = (const float*)d_in[16];
    const float* nb1    = (const float*)d_in[17];
    const float* nw2    = (const float*)d_in[18];
    const float* nb2    = (const float*)d_in[19];
    const float* cscal  = (const float*)d_in[20];

    const int N = in_sizes[0] / 64;
    const int E = in_sizes[3] / 2;
    const int* srcI = eidx;
    const int* dstI = eidx + E;

    char* ws = (char*)d_ws;
    size_t oH16  = 0;
    size_t oW1   = align256((size_t)N * 64 * 2);
    size_t oW2   = oW1  + 128 * 128 * 2;
    size_t oNW1  = oW2  + 128 * 64 * 2;
    size_t oNW2  = oNW1 + 128 * 128 * 2;
    size_t oVW1  = oNW2 + 128 * 64 * 2;
    size_t oMI   = align256(oVW1 + 64 * 128 * 2);
    size_t oVU   = oMI + (size_t)N * 64 * 4;

    _Float16* h16   = (_Float16*)(ws + oH16);
    _Float16* w1sw  = (_Float16*)(ws + oW1);
    _Float16* w2sw  = (_Float16*)(ws + oW2);
    _Float16* nw1sw = (_Float16*)(ws + oNW1);
    _Float16* nw2sw = (_Float16*)(ws + oNW2);
    _Float16* vw1sw = (_Float16*)(ws + oVW1);
    float*    m_i   = (float*)(ws + oMI);
    float*    v_upd = (float*)(ws + oVU);

    float* out = (float*)d_out;

    zero_f32_kernel<<<(N * 64 + 255) / 256, 256, 0, stream>>>(m_i, N * 64);
    zero_f32_kernel<<<(N * 3 + 255) / 256, 256, 0, stream>>>(v_upd, N * 3);
    cvt_f32_to_f16_kernel<<<(N * 64 + 255) / 256, 256, 0, stream>>>(h, h16, N * 64);

    swizzleB_kernel<<<32, 32, 0, stream>>>(ew1, w1sw, 128, 128);  // rows 0..127 of ew1
    swizzleB_kernel<<<16, 32, 0, stream>>>(ew2, w2sw, 128, 64);
    swizzleB_kernel<<<32, 32, 0, stream>>>(nw1, nw1sw, 128, 128);
    swizzleB_kernel<<<16, 32, 0, stream>>>(nw2, nw2sw, 128, 64);
    swizzleB_kernel<<<16, 32, 0, stream>>>(vw1, vw1sw, 64, 128);

    egnn_edge_kernel<<<(E + 255) / 256, 64, 0, stream>>>(
        h16, x, srcI, dstI, w1sw, w2sw, ew1, eb1, eb2,
        cw1, cb1, cw2, cb2, cscal, m_i, v_upd, E, N);

    egnn_node_h_kernel<<<(N + 255) / 256, 64, 0, stream>>>(
        h16, m_i, h, nw1sw, nw2sw, nb1, nb2, out, N);

    egnn_node_v_kernel<<<(N + 255) / 256, 64, 0, stream>>>(
        h16, x, v_init, vw2, vb2, vw1sw, vb1, v_upd, out, N);
}
